// SigLIPSlotHybrid_45672682225773
// MI455X (gfx1250) — matrix operand order repaired and checked by reading, not compiled, not run on hardware
//
#include <hip/hip_runtime.h>
#include <hip/hip_bf16.h>

// ---------------- model constants ----------------
#define B_    64
#define N_    1024
#define FD_   768
#define D_    256
#define KS_   8
#define TS_   12
#define H_    4
#define HD_   64
#define ITERS_ 3
#define L_    2
#define SCALE_ 0.125f   // HD^-0.5

// ---------------- types ----------------
typedef __bf16  v16bf __attribute__((ext_vector_type(16)));
typedef float   v8f   __attribute__((ext_vector_type(8)));
typedef unsigned int u32x4 __attribute__((ext_vector_type(4)));
typedef unsigned int u32x2 __attribute__((ext_vector_type(2)));
typedef float   f32x4 __attribute__((ext_vector_type(4)));

__device__ __forceinline__ unsigned short f2bf(float f) {
  unsigned int u = __float_as_uint(f);
  u += 0x7FFFu + ((u >> 16) & 1u);          // round-to-nearest-even
  return (unsigned short)(u >> 16);
}
__device__ __forceinline__ float gelu_exact(float x) {
  return 0.5f * x * (1.0f + erff(x * 0.70710678118654752f));
}

// ---------------- WMMA GEMM ----------------
// C[m,n] = epilogue( sum_k A[m,k] * B[k,n] ), bf16 multiply, f32 accumulate.
// Block tile 32x64, 8 waves (2x4 grid of 16x16 WMMA tiles).
// K step 64 -> 2 WMMAs per stage; double-buffered LDS -> 1 barrier per stage,
// next stage's global loads staged in registers during compute.
// Requires N % 64 == 0 and K % 64 == 0. M arbitrary (guarded).
// Batched: grid.z = nbatch, z decomposed as (zb, zh) with zh in [0, bh).
// op: 0=none 1=bias 2=bias+gelu 3=bias+residual 4=k-transpose scatter (aux=Ntok)
#define BM 32
#define BN 64
#define BK 64
#define KPAD 80   // 64 + 16; row stride = 160 B (16B aligned)

template<bool AF32, bool BF32>
__global__ __launch_bounds__(256) void gemm_wmma(
    const void* __restrict__ Av, const void* __restrict__ Bv,
    float* __restrict__ Cf, unsigned short* __restrict__ Cbf,
    int M, int N, int K, int lda, int ldb, int ldc,
    int bh, long sAb, long sAh, long sBb, long sBh, long sCb, long sCh,
    const float* __restrict__ bias,
    const float* __restrict__ resid, int ldres,
    int op, int aux)
{
  __shared__ __align__(16) unsigned short As[2][BM * KPAD];
  __shared__ __align__(16) unsigned short Bs[2][BN * KPAD];

  const int z  = blockIdx.z;
  const int zb = z / bh, zh = z % bh;
  const long offA = (long)zb * sAb + (long)zh * sAh;
  const long offB = (long)zb * sBb + (long)zh * sBh;
  const long offC = (long)zb * sCb + (long)zh * sCh;

  const int t    = threadIdx.x;
  const int lane = t & 31;
  const int wave = t >> 5;
  const int wm   = wave >> 2;   // 0..1
  const int wn   = wave & 3;    // 0..3
  const int blockM = blockIdx.y * BM;
  const int blockN = blockIdx.x * BN;

  // A tile: each thread loads 8 consecutive k of one row (32 rows x 8 thr/row)
  const int a_row = t >> 3;          // 0..31
  const int a_col = (t & 7) * 8;     // 0..56
  // B tile: each thread loads 16 consecutive n of one k (64 k x 4 thr/row)
  const int b_k = t >> 2;            // 0..63
  const int b_n = (t & 3) * 16;      // 0..48

  unsigned short areg[8];
  unsigned short breg[16];

  const int nsteps = K / BK;

  auto load_a = [&](int s) {
    const int mg = blockM + a_row;
    const int kg = s * BK + a_col;
    if (mg < M) {
      if (AF32) {
        const float* ap = (const float*)Av + offA + (long)mg * lda + kg;
        f32x4 f0 = *(const f32x4*)ap;
        f32x4 f1 = *(const f32x4*)(ap + 4);
        #pragma unroll
        for (int j = 0; j < 4; ++j) { areg[j] = f2bf(f0[j]); areg[4 + j] = f2bf(f1[j]); }
        __builtin_prefetch(ap + BK, 0, 3);
      } else {
        const unsigned short* ap = (const unsigned short*)Av + offA + (long)mg * lda + kg;
        union { u32x4 q; unsigned short s8[8]; } u;
        u.q = *(const u32x4*)ap;
        #pragma unroll
        for (int j = 0; j < 8; ++j) areg[j] = u.s8[j];
        __builtin_prefetch(ap + BK, 0, 3);
      }
    } else {
      #pragma unroll
      for (int j = 0; j < 8; ++j) areg[j] = 0;
    }
  };
  auto store_a = [&](int buf) {
    union { u32x4 q; unsigned short s8[8]; } u;
    #pragma unroll
    for (int j = 0; j < 8; ++j) u.s8[j] = areg[j];
    *(u32x4*)&As[buf][a_row * KPAD + a_col] = u.q;
  };
  auto load_b = [&](int s) {
    const int kg = s * BK + b_k;
    const int ng = blockN + b_n;
    if (BF32) {
      const float* bp = (const float*)Bv + offB + (long)kg * ldb + ng;
      f32x4 f0 = *(const f32x4*)bp;
      f32x4 f1 = *(const f32x4*)(bp + 4);
      f32x4 f2 = *(const f32x4*)(bp + 8);
      f32x4 f3 = *(const f32x4*)(bp + 12);
      #pragma unroll
      for (int j = 0; j < 4; ++j) {
        breg[j]      = f2bf(f0[j]);
        breg[4 + j]  = f2bf(f1[j]);
        breg[8 + j]  = f2bf(f2[j]);
        breg[12 + j] = f2bf(f3[j]);
      }
      __builtin_prefetch(bp + (long)BK * ldb, 0, 3);
    } else {
      const unsigned short* bp = (const unsigned short*)Bv + offB + (long)kg * ldb + ng;
      union { u32x4 q; unsigned short s8[8]; } u0, u1;
      u0.q = *(const u32x4*)bp;
      u1.q = *(const u32x4*)(bp + 8);
      #pragma unroll
      for (int j = 0; j < 8; ++j) { breg[j] = u0.s8[j]; breg[8 + j] = u1.s8[j]; }
      __builtin_prefetch(bp + (long)BK * ldb, 0, 3);
    }
  };
  auto store_b = [&](int buf) {
    #pragma unroll
    for (int j = 0; j < 16; ++j) Bs[buf][(b_n + j) * KPAD + b_k] = breg[j];
  };

  v8f acc = {0.f,0.f,0.f,0.f,0.f,0.f,0.f,0.f};
  const int arow  = wm * 16 + (lane & 15);
  const int koffA = (lane >> 4) * 8;
  const int brow  = wn * 16 + (lane & 15);
  const int koffB = (lane >> 4) * 8;

  // prologue: stage 0 into buffer 0
  load_a(0); load_b(0);
  store_a(0); store_b(0);

  for (int s = 0; s < nsteps; ++s) {
    __syncthreads();
    const int cur = s & 1;
    const bool more = (s + 1) < nsteps;
    if (more) { load_a(s + 1); load_b(s + 1); }
    // ---- 2 WMMAs on the current LDS stage ----
    #pragma unroll
    for (int kk = 0; kk < BK; kk += 32) {
      union { v16bf v; u32x4 q[2]; } fa, fb;
      fa.q[0] = *(const u32x4*)&As[cur][arow * KPAD + kk + koffA];
      fa.q[1] = *(const u32x4*)&As[cur][arow * KPAD + kk + koffA + 16];
      fb.q[0] = *(const u32x4*)&Bs[cur][brow * KPAD + kk + koffB];
      fb.q[1] = *(const u32x4*)&Bs[cur][brow * KPAD + kk + koffB + 16];
      acc = __builtin_amdgcn_wmma_f32_16x16x32_bf16(
          false, fa.v, false, fb.v, (short)0, acc, false, false);
    }
    if (more) { store_a(cur ^ 1); store_b(cur ^ 1); }
  }

  // ---- epilogue: VGPR r -> M = r + (lane>=16 ? 8:0), N = lane&15 ----
  const int mbase = blockM + wm * 16 + ((lane >> 4) << 3);
  const int nbase = blockN + wn * 16;
  const int n_l = lane & 15;
  #pragma unroll
  for (int r = 0; r < 8; ++r) {
    const int mg = mbase + r;
    const int ng = nbase + n_l;
    if (mg >= M || ng >= N) continue;
    float val = acc[r];
    if (op == 1 || op == 2 || op == 3) val += bias[ng];
    if (op == 2) val = gelu_exact(val);
    if (op == 3) val += resid[(long)mg * ldres + ng];
    if (op == 4) {
      // scatter k[(b*Ntok+nt), h*HD+hd] -> kt[b][h][hd][nt], aux = Ntok
      const int bb = mg / aux, nt = mg % aux;
      const int h = ng >> 6, hd = ng & 63;
      const long idx = (((long)(bb * H_ + h) * HD_ + hd)) * aux + nt;
      if (Cbf) Cbf[idx] = f2bf(val);
    } else {
      const long idx = offC + (long)mg * ldc + ng;
      if (Cf)  Cf[idx]  = val;
      if (Cbf) Cbf[idx] = f2bf(val);
    }
  }
}

// ---------------- LayerNorm (optionally applied twice) ----------------
__device__ __forceinline__ float block_sum_256(float v, float* red, int t) {
  red[t] = v; __syncthreads();
  #pragma unroll
  for (int o = 128; o > 0; o >>= 1) { if (t < o) red[t] += red[t + o]; __syncthreads(); }
  float r = red[0]; __syncthreads();
  return r;
}

__global__ __launch_bounds__(256) void ln_kernel(
    const float* __restrict__ in,
    const float* __restrict__ g1, const float* __restrict__ b1,
    const float* __restrict__ g2, const float* __restrict__ b2,
    float* __restrict__ outf, unsigned short* __restrict__ outbf)
{
  __shared__ float red[256];
  const int row = blockIdx.x, t = threadIdx.x;
  float x = in[(long)row * D_ + t];
  float m = block_sum_256(x, red, t) * (1.0f / D_);
  float d = x - m;
  float v = block_sum_256(d * d, red, t) * (1.0f / D_);
  float y = d * rsqrtf(v + 1e-5f) * g1[t] + b1[t];
  if (g2) {
    float m2 = block_sum_256(y, red, t) * (1.0f / D_);
    float d2 = y - m2;
    float v2 = block_sum_256(d2 * d2, red, t) * (1.0f / D_);
    y = d2 * rsqrtf(v2 + 1e-5f) * g2[t] + b2[t];
  }
  if (outf)  outf [(long)row * D_ + t] = y;
  if (outbf) outbf[(long)row * D_ + t] = f2bf(y);
}

// ---------------- slots init ----------------
__global__ void slots_init_k(const float* __restrict__ mu, const float* __restrict__ lsig,
                             const float* __restrict__ eps, float* __restrict__ slots)
{
  const int id = blockIdx.x * 256 + threadIdx.x;      // B*TS*D
  const int sd = id % (TS_ * D_);
  slots[id] = mu[sd] + expf(lsig[sd]) * eps[id];
}

// ---------------- softmax over the 48 (slot,head) values per (b,n) ----------------
__global__ __launch_bounds__(256) void softmax48_k(float* __restrict__ dots)
{
  const int id = blockIdx.x * 256 + threadIdx.x;  // B*N
  const int b = id >> 10, n = id & (N_ - 1);
  float v[H_ * TS_];
  float mx = -3.4e38f;
  #pragma unroll
  for (int h = 0; h < H_; ++h)
    #pragma unroll
    for (int i = 0; i < TS_; ++i) {
      float x = dots[((long)((b * H_ + h) * TS_ + i)) * N_ + n] * SCALE_;
      v[h * TS_ + i] = x; mx = fmaxf(mx, x);
    }
  float s = 0.f;
  #pragma unroll
  for (int j = 0; j < H_ * TS_; ++j) { v[j] = expf(v[j] - mx); s += v[j]; }
  const float inv = 1.f / s;
  #pragma unroll
  for (int h = 0; h < H_; ++h)
    #pragma unroll
    for (int i = 0; i < TS_; ++i)
      dots[((long)((b * H_ + h) * TS_ + i)) * N_ + n] = v[h * TS_ + i] * inv;
}

// ---------------- per-(b,h,slot) normalization over N ----------------
__global__ __launch_bounds__(256) void attn_norm_k(float* __restrict__ attn)
{
  __shared__ float red[256];
  const long base = (long)blockIdx.x * N_;
  const int t = threadIdx.x;
  float s = 0.f;
  for (int j = t; j < N_; j += 256) s += attn[base + j];
  s = block_sum_256(s, red, t);
  const float inv = 1.f / (s + 1e-8f);
  for (int j = t; j < N_; j += 256) attn[base + j] = (attn[base + j] + 1e-8f) * inv;
}

// ---------------- GRU gate math (in-place slots update) ----------------
__global__ __launch_bounds__(256) void gru_k(const float* __restrict__ gx,
                                             const float* __restrict__ gh,
                                             float* __restrict__ slots)
{
  const int id = blockIdx.x * 256 + threadIdx.x;   // (B*TS)*D
  const int m = id >> 8, c = id & (D_ - 1);
  const long b3 = (long)m * 3 * D_;
  const float xr = gx[b3 + c], xz = gx[b3 + D_ + c], xn = gx[b3 + 2 * D_ + c];
  const float hr = gh[b3 + c], hz = gh[b3 + D_ + c], hn = gh[b3 + 2 * D_ + c];
  const float r  = 1.f / (1.f + expf(-(xr + hr)));
  const float zz = 1.f / (1.f + expf(-(xz + hz)));
  const float nn = tanhf(xn + r * hn);
  const float h  = slots[id];
  slots[id] = (1.f - zz) * nn + zz * h;
}

// ---------------- selection head ----------------
__global__ __launch_bounds__(128) void sel_logits_k(
    const float* __restrict__ slots,
    const float* __restrict__ w1, const float* __restrict__ b1,
    const float* __restrict__ w2, const float* __restrict__ b2,
    float* __restrict__ dec)
{
  __shared__ float l0[128], l1[128];
  const int rk = blockIdx.x;                 // b*KS + k
  const int b = rk >> 3, k = rk & 7;
  const float* row = slots + (long)(b * TS_ + k) * D_;
  const int j = threadIdx.x;
  float h = b1[j];
  for (int c = 0; c < D_; ++c) h += row[c] * w1[c * (D_ / 2) + j];
  h = fmaxf(h, 0.f);
  l0[j] = h * w2[j * 2 + 0];
  l1[j] = h * w2[j * 2 + 1];
  __syncthreads();
  #pragma unroll
  for (int o = 64; o > 0; o >>= 1) {
    if (j < o) { l0[j] += l0[j + o]; l1[j] += l1[j + o]; }
    __syncthreads();
  }
  if (j == 0) dec[rk] = ((l1[0] + b2[1]) > (l0[0] + b2[0])) ? 1.f : 0.f;
}

__global__ void sel_fix_k(float* __restrict__ dec)
{
  if (threadIdx.x != 0) return;
  const int b = blockIdx.x;
  float d[KS_]; float s = 0.f;
  for (int k = 0; k < KS_; ++k) { d[k] = dec[b * KS_ + k]; s += d[k]; }
  const float needed = fmaxf(2.f - s, 0.f);
  float c = 0.f;
  for (int k = 0; k < KS_; ++k) {
    const float ina = 1.f - d[k];
    c += ina;
    if (ina > 0.f && c <= needed) d[k] = 1.f;
    dec[b * KS_ + k] = d[k];
  }
}

__global__ __launch_bounds__(256) void xmask_k(const float* __restrict__ slots,
                                               const float* __restrict__ dec,
                                               float* __restrict__ x)
{
  const int rk = blockIdx.x, c = threadIdx.x;
  const int b = rk >> 3, k = rk & 7;
  x[(long)rk * D_ + c] = slots[(long)(b * TS_ + k) * D_ + c] * dec[rk];
}

// ---------------- tiny 8-node GAT attention, y = x + attn(hmat) ----------------
__global__ __launch_bounds__(64) void gat_attn_k(
    const float* __restrict__ hmat, const float* __restrict__ a,
    const float* __restrict__ dec, const float* __restrict__ xin,
    float* __restrict__ y)
{
  __shared__ float red[64];
  __shared__ float ei[KS_], ej[KS_];
  __shared__ float alpha[KS_][KS_];
  const int z = blockIdx.x;           // b*H + h
  const int b = z >> 2, h = z & 3;
  const int t = threadIdx.x;          // hd
  const float* a1 = a + h * 2 * HD_;
  const float* a2 = a1 + HD_;
  for (int k = 0; k < KS_; ++k) {
    const float hv = hmat[(long)(b * KS_ + k) * D_ + h * HD_ + t];
    red[t] = hv * a1[t]; __syncthreads();
    #pragma unroll
    for (int o = 32; o > 0; o >>= 1) { if (t < o) red[t] += red[t + o]; __syncthreads(); }
    if (t == 0) ei[k] = red[0];
    __syncthreads();
    red[t] = hv * a2[t]; __syncthreads();
    #pragma unroll
    for (int o = 32; o > 0; o >>= 1) { if (t < o) red[t] += red[t + o]; __syncthreads(); }
    if (t == 0) ej[k] = red[0];
    __syncthreads();
  }
  if (t < KS_) {
    const float dk = dec[b * KS_ + t];
    float ev[KS_]; float mx = -3.4e38f;
    for (int n = 0; n < KS_; ++n) {
      float e = ei[t] + ej[n];
      e = (e >= 0.f) ? e : 0.2f * e;                 // LeakyReLU
      if (dk * dec[b * KS_ + n] == 0.f) e = -1e9f;   // mask
      ev[n] = e; mx = fmaxf(mx, e);
    }
    float s = 0.f;
    for (int n = 0; n < KS_; ++n) { ev[n] = expf(ev[n] - mx); s += ev[n]; }
    for (int n = 0; n < KS_; ++n) alpha[t][n] = ev[n] / s;
  }
  __syncthreads();
  for (int k = 0; k < KS_; ++k) {
    float acc = 0.f;
    for (int n = 0; n < KS_; ++n)
      acc += alpha[k][n] * hmat[(long)(b * KS_ + n) * D_ + h * HD_ + t];
    const long o = (long)(b * KS_ + k) * D_ + h * HD_ + t;
    y[o] = xin[o] + acc;
  }
}

// ---------------- host orchestration ----------------
extern "C" void kernel_launch(void* const* d_in, const int* in_sizes, int n_in,
                              void* d_out, int out_size, void* d_ws, size_t ws_size,
                              hipStream_t stream)
{
  (void)in_sizes; (void)n_in; (void)out_size; (void)ws_size;
  const float* features       = (const float*)d_in[0];
  const float* eps_noise      = (const float*)d_in[1];
  const float* in_w           = (const float*)d_in[2];
  const float* in_b           = (const float*)d_in[3];
  const float* in_ln_g        = (const float*)d_in[4];
  const float* in_ln_b        = (const float*)d_in[5];
  const float* ni_g           = (const float*)d_in[6];
  const float* ni_b           = (const float*)d_in[7];
  const float* slot_mu        = (const float*)d_in[8];
  const float* slot_log_sigma = (const float*)d_in[9];
  const float* q_w            = (const float*)d_in[10];
  const float* k_w            = (const float*)d_in[11];
  const float* v_w            = (const float*)d_in[12];
  const float* gru_wih        = (const float*)d_in[13];
  const float* gru_whh        = (const float*)d_in[14];
  const float* gru_bih        = (const float*)d_in[15];
  const float* gru_bhh        = (const float*)d_in[16];
  const float* ns_g           = (const float*)d_in[17];
  const float* ns_b           = (const float*)d_in[18];
  const float* ffn_ln_g       = (const float*)d_in[19];
  const float* ffn_ln_b       = (const float*)d_in[20];
  const float* ffn_w1         = (const float*)d_in[21];
  const float* ffn_b1         = (const float*)d_in[22];
  const float* ffn_w2         = (const float*)d_in[23];
  const float* ffn_b2         = (const float*)d_in[24];
  const float* sel_w1         = (const float*)d_in[25];
  const float* sel_b1         = (const float*)d_in[26];
  const float* sel_w2         = (const float*)d_in[27];
  const float* sel_b2         = (const float*)d_in[28];
  const float* gat_W          = (const float*)d_in[29];
  const float* gat_a          = (const float*)d_in[30];
  const float* gat_ln_g       = (const float*)d_in[31];
  const float* gat_ln_b       = (const float*)d_in[32];
  const float* gat_fln_g      = (const float*)d_in[33];
  const float* gat_fln_b      = (const float*)d_in[34];
  const float* gat_fw1        = (const float*)d_in[35];
  const float* gat_fb1        = (const float*)d_in[36];
  const float* gat_fw2        = (const float*)d_in[37];
  const float* gat_fb2        = (const float*)d_in[38];

  // ---- workspace layout ----
  char* ws = (char*)d_ws;
  size_t off = 0;
  auto alloc = [&](size_t bytes) -> void* {
    void* p = ws + off; off += (bytes + 255) & ~(size_t)255; return p;
  };
  const int MT = B_ * N_;                              // 65536 tokens
  float*          tmp0    = (float*)         alloc((size_t)MT * D_ * 4);   // pre-LN; reused as kt
  unsigned short* xn_bf   = (unsigned short*)alloc((size_t)MT * D_ * 2);
  unsigned short* v_bf    = (unsigned short*)alloc((size_t)MT * D_ * 2);
  float*          dots    = (float*)         alloc((size_t)B_ * H_ * TS_ * N_ * 4);
  float*          qbuf    = (float*)         alloc((size_t)B_ * TS_ * D_ * 4);
  unsigned short* sn_bf   = (unsigned short*)alloc((size_t)B_ * TS_ * D_ * 2);
  float*          updates = (float*)         alloc((size_t)B_ * TS_ * D_ * 4);
  float*          gx      = (float*)         alloc((size_t)B_ * TS_ * 3 * D_ * 4);
  float*          gh      = (float*)         alloc((size_t)B_ * TS_ * 3 * D_ * 4);
  float*          slots   = (float*)         alloc((size_t)B_ * TS_ * D_ * 4);
  unsigned short* fbf     = (unsigned short*)alloc((size_t)B_ * TS_ * D_ * 2);
  unsigned short* h1_bf   = (unsigned short*)alloc((size_t)B_ * TS_ * 4 * D_ * 2);
  float*          dec     = (float*)         alloc((size_t)B_ * KS_ * 4);
  float*          xg      = (float*)         alloc((size_t)B_ * KS_ * D_ * 4);
  float*          hmat    = (float*)         alloc((size_t)B_ * KS_ * D_ * 4);
  float*          ybuf    = (float*)         alloc((size_t)B_ * KS_ * D_ * 4);
  unsigned short* g1_bf   = (unsigned short*)alloc((size_t)B_ * KS_ * 4 * D_ * 2);
  unsigned short* kt      = (unsigned short*)tmp0;     // aliased after ln consumed tmp0

  const dim3 blk(256);

  // 1) tmp0 = features @ in_w + in_b          (M=65536, K=768, N=256)
  {
    dim3 g(D_ / BN, MT / BM, 1);
    gemm_wmma<true, true><<<g, blk, 0, stream>>>(features, in_w, tmp0, nullptr,
        MT, D_, FD_, FD_, D_, D_, 1, 0,0,0,0,0,0, in_b, nullptr, 0, 1, 0);
  }
  // 2) x_n = LN(LN(tmp0))  -> bf16
  ln_kernel<<<MT, blk, 0, stream>>>(tmp0, in_ln_g, in_ln_b, ni_g, ni_b, nullptr, xn_bf);
  // 3) kt[b][h][hd][n] = (x_n @ k_w) scattered (bf16), aliased into tmp0
  {
    dim3 g(D_ / BN, MT / BM, 1);
    gemm_wmma<false, true><<<g, blk, 0, stream>>>(xn_bf, k_w, nullptr, kt,
        MT, D_, D_, D_, D_, 0, 1, 0,0,0,0,0,0, nullptr, nullptr, 0, 4, N_);
  }
  // 4) v = x_n @ v_w  -> bf16 token-major
  {
    dim3 g(D_ / BN, MT / BM, 1);
    gemm_wmma<false, true><<<g, blk, 0, stream>>>(xn_bf, v_w, nullptr, v_bf,
        MT, D_, D_, D_, D_, D_, 1, 0,0,0,0,0,0, nullptr, nullptr, 0, 0, 0);
  }
  // 5) slots = mu + exp(log_sigma) * eps
  slots_init_k<<<(B_ * TS_ * D_) / 256, blk, 0, stream>>>(slot_mu, slot_log_sigma, eps_noise, slots);

  const int MS = B_ * TS_;   // 768 slot rows
  for (int it = 0; it < ITERS_; ++it) {
    // sn = LN(slots) -> bf16
    ln_kernel<<<MS, blk, 0, stream>>>(slots, ns_g, ns_b, nullptr, nullptr, nullptr, sn_bf);
    // q = sn @ q_w
    {
      dim3 g(D_ / BN, MS / BM, 1);
      gemm_wmma<false, true><<<g, blk, 0, stream>>>(sn_bf, q_w, qbuf, nullptr,
          MS, D_, D_, D_, D_, D_, 1, 0,0,0,0,0,0, nullptr, nullptr, 0, 0, 0);
    }
    // dots[b,h,i,n] = q[b,i,h,:] . kt[b,h,:,n]      (batched over b,h)
    {
      dim3 g(N_ / BN, 1, B_ * H_);
      gemm_wmma<true, false><<<g, blk, 0, stream>>>(qbuf, kt, dots, nullptr,
          TS_, N_, HD_, D_, N_, N_,
          H_, (long)TS_ * D_, (long)HD_,
              (long)H_ * HD_ * N_, (long)HD_ * N_,
              (long)H_ * TS_ * N_, (long)TS_ * N_,
          nullptr, nullptr, 0, 0, 0);
    }
    softmax48_k<<<(B_ * N_) / 256, blk, 0, stream>>>(dots);
    attn_norm_k<<<B_ * H_ * TS_, blk, 0, stream>>>(dots);
    // updates[b,i,h,:] = attn_n[b,h,i,:] @ v[b,:,h,:]
    {
      dim3 g(HD_ / BN, 1, B_ * H_);
      gemm_wmma<true, false><<<g, blk, 0, stream>>>(dots, v_bf, updates, nullptr,
          TS_, HD_, N_, N_, D_, D_,
          H_, (long)H_ * TS_ * N_, (long)TS_ * N_,
              (long)N_ * D_, (long)HD_,
              (long)TS_ * D_, (long)HD_,
          nullptr, nullptr, 0, 0, 0);
    }
    // GRU
    {
      dim3 g((3 * D_) / BN, MS / BM, 1);
      gemm_wmma<true, true><<<g, blk, 0, stream>>>(updates, gru_wih, gx, nullptr,
          MS, 3 * D_, D_, D_, 3 * D_, 3 * D_, 1, 0,0,0,0,0,0, gru_bih, nullptr, 0, 1, 0);
      gemm_wmma<true, true><<<g, blk, 0, stream>>>(slots, gru_whh, gh, nullptr,
          MS, 3 * D_, D_, D_, 3 * D_, 3 * D_, 1, 0,0,0,0,0,0, gru_bhh, nullptr, 0, 1, 0);
    }
    gru_k<<<(MS * D_) / 256, blk, 0, stream>>>(gx, gh, slots);
    // FFN with residual
    ln_kernel<<<MS, blk, 0, stream>>>(slots, ffn_ln_g, ffn_ln_b, nullptr, nullptr, nullptr, fbf);
    {
      dim3 g((4 * D_) / BN, MS / BM, 1);
      gemm_wmma<false, true><<<g, blk, 0, stream>>>(fbf, ffn_w1, nullptr, h1_bf,
          MS, 4 * D_, D_, D_, 4 * D_, 4 * D_, 1, 0,0,0,0,0,0, ffn_b1, nullptr, 0, 2, 0);
    }
    {
      dim3 g(D_ / BN, MS / BM, 1);
      gemm_wmma<false, true><<<g, blk, 0, stream>>>(h1_bf, ffn_w2, slots, nullptr,
          MS, D_, 4 * D_, 4 * D_, D_, D_, 1, 0,0,0,0,0,0, ffn_b2, slots, D_, 3, 0);
    }
  }

  // selection head + decision fixup + masked objects
  sel_logits_k<<<B_ * KS_, dim3(128), 0, stream>>>(slots, sel_w1, sel_b1, sel_w2, sel_b2, dec);
  sel_fix_k<<<B_, dim3(32), 0, stream>>>(dec);
  xmask_k<<<B_ * KS_, blk, 0, stream>>>(slots, dec, xg);

  // GAT layers
  const int MG = B_ * KS_;   // 512 rows
  for (int l = 0; l < L_; ++l) {
    ln_kernel<<<MG, blk, 0, stream>>>(xg, gat_ln_g + l * D_, gat_ln_b + l * D_,
                                      nullptr, nullptr, nullptr, sn_bf);
    {
      dim3 g(D_ / BN, MG / BM, 1);
      gemm_wmma<false, true><<<g, blk, 0, stream>>>(sn_bf, gat_W + (size_t)l * D_ * D_,
          hmat, nullptr, MG, D_, D_, D_, D_, D_, 1, 0,0,0,0,0,0, nullptr, nullptr, 0, 0, 0);
    }
    gat_attn_k<<<B_ * H_, dim3(64), 0, stream>>>(hmat, gat_a + (size_t)l * H_ * 2 * HD_,
                                                 dec, xg, ybuf);
    ln_kernel<<<MG, blk, 0, stream>>>(ybuf, gat_fln_g + l * D_, gat_fln_b + l * D_,
                                      nullptr, nullptr, nullptr, fbf);
    {
      dim3 g((4 * D_) / BN, MG / BM, 1);
      gemm_wmma<false, true><<<g, blk, 0, stream>>>(fbf, gat_fw1 + (size_t)l * D_ * 4 * D_,
          nullptr, g1_bf, MG, 4 * D_, D_, D_, 4 * D_, 4 * D_, 1, 0,0,0,0,0,0,
          gat_fb1 + l * 4 * D_, nullptr, 0, 2, 0);
    }
    {
      float* Cdst = (l == L_ - 1) ? (float*)d_out : xg;
      dim3 g(D_ / BN, MG / BM, 1);
      gemm_wmma<false, true><<<g, blk, 0, stream>>>(g1_bf, gat_fw2 + (size_t)l * 4 * D_ * D_,
          Cdst, nullptr, MG, D_, 4 * D_, 4 * D_, D_, D_, 1, 0,0,0,0,0,0,
          gat_fb2 + l * D_, ybuf, D_, 3, 0);
    }
  }
}
